// Attention_24842090840644
// MI455X (gfx1250) — matrix operand order repaired and checked by reading, not compiled, not run on hardware
//
#include <hip/hip_runtime.h>
#include <hip/hip_bf16.h>

#define BATCH 64
#define TLEN 2048
#define D_LF 512
#define D_DEC 256
#define D_KQ 128
#define D_CTX 128

typedef __attribute__((ext_vector_type(16))) __bf16 v16bf;
typedef __attribute__((ext_vector_type(8)))  float  v8f;
typedef __attribute__((ext_vector_type(8)))  unsigned int v8u;

struct U8x32 { uint4 a, b; };   // 32 bytes -> bit_cast to v16bf

static __device__ __forceinline__ unsigned int cvt2(float lo, float hi) {
    // native bf16 converts; compiler can fuse into v_cvt_pk_bf16_f32
    unsigned short l = __builtin_bit_cast(unsigned short, (__bf16)lo);
    unsigned short h = __builtin_bit_cast(unsigned short, (__bf16)hi);
    return (unsigned int)l | ((unsigned int)h << 16);
}
static __device__ __forceinline__ unsigned short f32_to_bf16(float f) {
    return __builtin_bit_cast(unsigned short, (__bf16)f);
}
static __device__ __forceinline__ float bf16_to_f32(unsigned short h) {
    return __uint_as_float(((unsigned int)h) << 16);
}
static __device__ __forceinline__ float lrelu(float x) {
    return x >= 0.f ? x : 0.2f * x;
}

// ---------------- Kernel 0: convert Wk / Wv (fp32 [128,512]) to bf16 copies --
__global__ void cvt_w_kernel(const float* __restrict__ Wk, const float* __restrict__ Wv,
                             unsigned short* __restrict__ wkb, unsigned short* __restrict__ wvb) {
    int i = blockIdx.x * 256 + threadIdx.x;   // 65536 elements each
    wkb[i] = f32_to_bf16(Wk[i]);
    wvb[i] = f32_to_bf16(Wv[i]);
}

// ---------------- Kernel 1: query = LReLU(ds @ Wq^T + bq)  [64,128] ----------
__global__ void query_kernel(const float* __restrict__ ds, const float* __restrict__ Wq,
                             const float* __restrict__ bq, float* __restrict__ qws) {
    const int b = blockIdx.x;
    const int n = threadIdx.x;                // 128 threads
    const float* x = ds + (size_t)b * D_DEC;
    const float* w = Wq + (size_t)n * D_DEC;
    float s = bq[n];
    for (int k = 0; k < D_DEC; ++k) s = fmaf(x[k], w[k], s);
    qws[b * D_KQ + n] = lrelu(s);
}

// ---------------- Kernel 2: fused key/value GEMMs + energy (WMMA bf16) -------
// grid: (T/32, B), block 256 (8 waves). Wave w owns N-tile [16w,16w+16);
// each wave computes two 16x16 M-tiles (rows t0..t0+15 and t0+16..t0+31).
#define APAD 8
#define ASTR (D_LF + APAD)      // 520 ushorts per row; 1040B stride (bank-safe)

__global__ __launch_bounds__(256) void kv_energy_kernel(
    const float* __restrict__ lf,
    const unsigned short* __restrict__ wk, const unsigned short* __restrict__ wv,
    const float* __restrict__ qws,
    const float* __restrict__ bk, const float* __restrict__ bv,
    float* __restrict__ ews, unsigned short* __restrict__ vws) {

    __shared__ __align__(16) unsigned short aT[32][ASTR];   // bf16 A tile, 32 x 512
    __shared__ float e_sh[32];

    const int b   = blockIdx.y;
    const int t0  = blockIdx.x * 32;
    const int tid = threadIdx.x;
    if (tid < 32) e_sh[tid] = 0.f;

    // ---- stage A tile: 32 rows x 512 fp32 -> bf16 in LDS (convert ONCE) ----
    {
        const float* src = lf + ((size_t)b * TLEN + t0) * D_LF;
        #pragma unroll
        for (int i = tid; i < 32 * (D_LF / 4); i += 256) {      // float4 granularity
            const int row = i >> 7;            // 128 float4 per row
            const int c4  = i & 127;
            float4 v = reinterpret_cast<const float4*>(src + (size_t)row * D_LF)[c4];
            uint2 p; p.x = cvt2(v.x, v.y); p.y = cvt2(v.z, v.w);
            *reinterpret_cast<uint2*>(&aT[row][c4 * 4]) = p;
        }
    }
    __syncthreads();

    const int wid  = tid >> 5;
    const int lane = tid & 31;
    const int half = lane >> 4;                 // K-half selector
    const int mrow = lane & 15;                 // A row within tile
    const int n    = wid * 16 + (lane & 15);    // B/C column

    const unsigned short* wkr = wk + (size_t)n * D_LF;   // weights in [N][K]
    const unsigned short* wvr = wv + (size_t)n * D_LF;
    const unsigned short* a0p = &aT[mrow][0];
    const unsigned short* a1p = &aT[mrow + 16][0];

    v8f ck0 = {}, ck1 = {}, cv0 = {}, cv1 = {};

    #pragma unroll 4
    for (int k0 = 0; k0 < D_LF; k0 += 32) {
        const int o0 = k0 + 16 * half;          // lanes 0-15: K {0..15}; 16-31: {16..31}
        const int o1 = k0 + 16 * half + 8;
        U8x32 u0, u1;
        u0.a = *reinterpret_cast<const uint4*>(a0p + o0);
        u0.b = *reinterpret_cast<const uint4*>(a0p + o1);
        u1.a = *reinterpret_cast<const uint4*>(a1p + o0);
        u1.b = *reinterpret_cast<const uint4*>(a1p + o1);
        v16bf A0 = __builtin_bit_cast(v16bf, u0);
        v16bf A1 = __builtin_bit_cast(v16bf, u1);

        const int kb = k0 + 16 * half;          // B: 16 contiguous bf16 of W[n][*]
        v16bf Bk = __builtin_bit_cast(v16bf, *reinterpret_cast<const v8u*>(wkr + kb));
        v16bf Bv = __builtin_bit_cast(v16bf, *reinterpret_cast<const v8u*>(wvr + kb));

        ck0 = __builtin_amdgcn_wmma_f32_16x16x32_bf16(false, A0, false, Bk, (short)0, ck0, false, false);
        ck1 = __builtin_amdgcn_wmma_f32_16x16x32_bf16(false, A1, false, Bk, (short)0, ck1, false, false);
        cv0 = __builtin_amdgcn_wmma_f32_16x16x32_bf16(false, A0, false, Bv, (short)0, cv0, false, false);
        cv1 = __builtin_amdgcn_wmma_f32_16x16x32_bf16(false, A1, false, Bv, (short)0, cv1, false, false);
    }

    // Epilogue: C layout -> lanes 0-15: M=r ; lanes 16-31: M=r+8 ; N = lane&15
    const float qn  = qws[b * D_KQ + n];
    const float bkn = bk[n];
    const float bvn = bv[n];
    unsigned short* vbase = vws + ((size_t)b * TLEN + t0) * D_CTX + n;
    #pragma unroll
    for (int r = 0; r < 8; ++r) {
        const int row0 = r + 8 * half;
        const int row1 = row0 + 16;
        float kx0 = lrelu(ck0[r] + bkn);
        float kx1 = lrelu(ck1[r] + bkn);
        atomicAdd(&e_sh[row0], kx0 * qn);        // ds_add_f32
        atomicAdd(&e_sh[row1], kx1 * qn);
        float vx0 = lrelu(cv0[r] + bvn);
        float vx1 = lrelu(cv1[r] + bvn);
        vbase[(size_t)row0 * D_CTX] = f32_to_bf16(vx0);
        vbase[(size_t)row1 * D_CTX] = f32_to_bf16(vx1);
    }
    __syncthreads();
    if (tid < 32) ews[(size_t)b * TLEN + t0 + tid] = e_sh[tid];
}

// ---------------- Kernel 3: stable softmax + mask + renormalize --------------
__global__ __launch_bounds__(256) void softmax_kernel(
    const float* __restrict__ ews, const int* __restrict__ seq, float* __restrict__ out) {
    __shared__ float row[TLEN];
    __shared__ float red[256];
    const int b = blockIdx.x, tid = threadIdx.x;
    const int S = seq[b];
    const float* e = ews + (size_t)b * TLEN;

    float m = -3.4e38f;
    for (int j = tid; j < TLEN; j += 256) { float v = e[j]; row[j] = v; m = fmaxf(m, v); }
    red[tid] = m; __syncthreads();
    for (int s = 128; s > 0; s >>= 1) { if (tid < s) red[tid] = fmaxf(red[tid], red[tid + s]); __syncthreads(); }
    const float mall = red[0]; __syncthreads();

    float sum = 0.f;
    for (int j = tid; j < TLEN; j += 256) {
        float ex = __expf(row[j] - mall);
        row[j] = ex;
        if (j < S) sum += ex;                    // masked sum (full-softmax denom cancels)
    }
    red[tid] = sum; __syncthreads();
    for (int s = 128; s > 0; s >>= 1) { if (tid < s) red[tid] += red[tid + s]; __syncthreads(); }
    const float inv = 1.f / red[0];

    for (int j = tid; j < TLEN; j += 256)
        out[(size_t)b * TLEN + j] = (j < S) ? row[j] * inv : 0.f;
}

// ---------------- Kernel 4a: partial context over t-chunks -------------------
#define NCHUNK 16
#define CH_T   (TLEN / NCHUNK)    // 128
__global__ __launch_bounds__(128) void context_partial_kernel(
    const float* __restrict__ score, const unsigned short* __restrict__ vws,
    float* __restrict__ part) {
    const int b = blockIdx.y, chunk = blockIdx.x;
    const int c = threadIdx.x;                  // 128 threads
    const float* s = score + (size_t)b * TLEN + chunk * CH_T;
    const unsigned short* v = vws + ((size_t)b * TLEN + (size_t)chunk * CH_T) * D_CTX + c;
    float acc = 0.f;
    #pragma unroll 4
    for (int t = 0; t < CH_T; ++t)
        acc = fmaf(s[t], bf16_to_f32(v[(size_t)t * D_CTX]), acc);
    part[((size_t)b * NCHUNK + chunk) * D_CTX + c] = acc;
}

// ---------------- Kernel 4b: reduce partials -> context ----------------------
__global__ __launch_bounds__(128) void context_reduce_kernel(
    const float* __restrict__ part, float* __restrict__ ctx) {
    const int b = blockIdx.x, c = threadIdx.x;
    float acc = 0.f;
    #pragma unroll
    for (int i = 0; i < NCHUNK; ++i) acc += part[((size_t)b * NCHUNK + i) * D_CTX + c];
    ctx[b * D_CTX + c] = acc;
}

extern "C" void kernel_launch(void* const* d_in, const int* in_sizes, int n_in,
                              void* d_out, int out_size, void* d_ws, size_t ws_size,
                              hipStream_t stream) {
    const float* ds  = (const float*)d_in[0];   // [64,256]
    const float* lf  = (const float*)d_in[1];   // [64,2048,512]
    const int*   seq = (const int*)  d_in[2];   // [64]
    const float* Wq  = (const float*)d_in[3];   // [128,256]
    const float* bq  = (const float*)d_in[4];   // [128]
    const float* Wk  = (const float*)d_in[5];   // [128,512]
    const float* bk  = (const float*)d_in[6];   // [128]
    const float* Wv  = (const float*)d_in[7];   // [128,512]
    const float* bv  = (const float*)d_in[8];   // [128]

    float* out_score = (float*)d_out;                       // [64,2048]
    float* out_ctx   = out_score + (size_t)BATCH * TLEN;    // [64,128]

    // workspace layout
    char* ws = (char*)d_ws;
    unsigned short* wkb  = (unsigned short*)(ws);                 // 128KB
    unsigned short* wvb  = (unsigned short*)(ws + 131072);        // 128KB
    float*          qws  = (float*)(ws + 262144);                 // 32KB
    float*          ews  = (float*)(ws + 294912);                 // 512KB
    float*          part = (float*)(ws + 819200);                 // 512KB
    unsigned short* vws  = (unsigned short*)(ws + 1343488);       // 32MB bf16 value

    cvt_w_kernel<<<dim3(D_KQ * D_LF / 256), dim3(256), 0, stream>>>(Wk, Wv, wkb, wvb);
    query_kernel<<<dim3(BATCH), dim3(D_KQ), 0, stream>>>(ds, Wq, bq, qws);
    kv_energy_kernel<<<dim3(TLEN / 32, BATCH), dim3(256), 0, stream>>>(
        lf, wkb, wvb, qws, bk, bv, ews, vws);
    softmax_kernel<<<dim3(BATCH), dim3(256), 0, stream>>>(ews, seq, out_score);
    context_partial_kernel<<<dim3(NCHUNK, BATCH), dim3(D_CTX), 0, stream>>>(out_score, vws, part);
    context_reduce_kernel<<<dim3(BATCH), dim3(D_CTX), 0, stream>>>(part, out_ctx);
}
